// DynamicGRU_55130200211745
// MI455X (gfx1250) — matrix operand order repaired and checked by reading, not compiled, not run on hardware
//
#include <hip/hip_runtime.h>
#include <hip/hip_bf16.h>

typedef __attribute__((ext_vector_type(16))) _Float16 v16h_t;
typedef __attribute__((ext_vector_type(8)))  float    v8f_t;

#define TT 200
#define DD 64
#define HH 64
#define GATES 192
#define NTILE 12   // 192 gate rows / 16 per WMMA M-tile

union Frag16 { v16h_t v; _Float16 h[16]; uint32_t u[8]; };

__device__ __forceinline__ float fast_sigmoid(float x) {
  // sigmoid(x) = 1 / (1 + 2^(-x*log2e));  large |x| saturates cleanly via rcp(inf)=0
  float e = __builtin_amdgcn_exp2f(-1.44269504f * x);
  return __builtin_amdgcn_rcpf(1.0f + e);
}
__device__ __forceinline__ float fast_tanh(float x) {
  // tanh(x) = 1 - 2/(2^(2x*log2e)+1);  saturates to +/-1 without NaN
  float e = __builtin_amdgcn_exp2f(2.88539008f * x);
  return 1.0f - 2.0f * __builtin_amdgcn_rcpf(e + 1.0f);
}

__global__ __launch_bounds__(32) void augru_scan_kernel(
    const float* __restrict__ x,    // (B,T,D)
    const float* __restrict__ att,  // (B,T)
    const int*   __restrict__ slen, // (B,)
    const float* __restrict__ w,    // (192,64)
    const float* __restrict__ u,    // (192,64)
    const float* __restrict__ bu, const float* __restrict__ br,
    const float* __restrict__ bh,
    float* __restrict__ out)        // (B,T,H)
{
  // Pre-swizzled A-matrix fragments (f16), 32B per lane per fragment, conflict-free b128 reads.
  __shared__ __align__(32) _Float16 wfr[NTILE * 2 * 32 * 16]; // 24 KB
  __shared__ __align__(32) _Float16 ufr[NTILE * 2 * 32 * 16]; // 24 KB
  __shared__ float bsh[GATES];                                // [bu|br|bh]

  const int lane = threadIdx.x & 31;
  const int ml   = lane & 15;       // batch column within tile / gate-row within M-tile
  const int hi   = lane >> 4;       // half-wave selector
  const int b    = blockIdx.x * 16 + ml;

  // ---- one-time: pack w,u into LDS in WMMA A-fragment layout (f32 -> f16) ----
  // A order used here (16x32 f16, same in B): lane L holds row M=L&15; half-element e maps to
  // K = e + 16*(L>>4) + 32*frag  -> one run of 16 consecutive K.
  for (int j = 0; j < NTILE; ++j) {
    const float* wrow = w + (16 * j + ml) * DD;
    const float* urow = u + (16 * j + ml) * HH;
    for (int f = 0; f < 2; ++f) {
      const int base = 32 * f + 16 * hi;
      _Float16* dw = &wfr[((j * 2 + f) * 32 + lane) * 16];
      _Float16* du = &ufr[((j * 2 + f) * 32 + lane) * 16];
      #pragma unroll
      for (int e = 0; e < 8; ++e) {
        dw[e]     = (_Float16)wrow[base + e];
        dw[8 + e] = (_Float16)wrow[base + 8 + e];
        du[e]     = (_Float16)urow[base + e];
        du[8 + e] = (_Float16)urow[base + 8 + e];
      }
    }
  }
  for (int i = lane; i < GATES; i += 32)
    bsh[i] = (i < 64) ? bu[i] : (i < 128) ? br[i - 64] : bh[i - 128];
  // Single-wave workgroup: DS ops are in-order; fence so reads see the packed data.
  asm volatile("s_wait_dscnt 0" ::: "memory");

  const float* xbase = x   + (size_t)b * TT * DD + 16 * hi;
  const float* abase = att + (size_t)b * TT;
  float*       obase = out + (size_t)b * TT * DD + 8 * hi;
  const int sl = slen[b];

  // h state in transposed-D layout: lane=batch col, hst[j][r] = h[b][16j + 8*hi + r]
  v8f_t hst[4] = {};
  Frag16 hb[2];                    // B-layout f16 fragments of h^T (K=H)
  #pragma unroll
  for (int f = 0; f < 2; ++f)
    #pragma unroll
    for (int q = 0; q < 8; ++q) hb[f].u[q] = 0u;

  for (int t = 0; t < TT; ++t) {
    const float* xt = xbase + t * DD;
    __builtin_prefetch(xbase + (t + 1) * DD, 0, 0);   // global_prefetch next step's x row

    // ---- B-fragments of x^T: lane holds 16 consecutive D at d0 = 16*hi + 32*f ----
    Frag16 xb[2];
    #pragma unroll
    for (int f = 0; f < 2; ++f) {
      const float* xp = xt + 32 * f;
      #pragma unroll
      for (int e = 0; e < 16; ++e) xb[f].h[e] = (_Float16)xp[e];
    }

    // ---- accumulators: fold biases in as C (broadcast LDS read) ----
    v8f_t acc[NTILE], acz[4];
    #pragma unroll
    for (int j = 0; j < NTILE; ++j) {
      const float* bp = &bsh[16 * j + 8 * hi];
      v8f_t a;
      #pragma unroll
      for (int r = 0; r < 8; ++r) a[r] = bp[r];
      acc[j] = a;
    }
    #pragma unroll
    for (int j = 0; j < 4; ++j) acz[j] = (v8f_t){};

    // ---- ix^T = w @ x^T (+bias) : 24 WMMAs ----
    #pragma unroll
    for (int j = 0; j < NTILE; ++j)
      #pragma unroll
      for (int f = 0; f < 2; ++f) {
        const Frag16* ap = (const Frag16*)&wfr[((j * 2 + f) * 32 + lane) * 16];
        acc[j] = __builtin_amdgcn_wmma_f32_16x16x32_f16(
            false, ap->v, false, xb[f].v, (short)0, acc[j], false, false);
      }
    // ---- hx^T = u @ h^T : u,r gates accumulate in place; z kept separate ----
    #pragma unroll
    for (int j = 0; j < 8; ++j)
      #pragma unroll
      for (int f = 0; f < 2; ++f) {
        const Frag16* ap = (const Frag16*)&ufr[((j * 2 + f) * 32 + lane) * 16];
        acc[j] = __builtin_amdgcn_wmma_f32_16x16x32_f16(
            false, ap->v, false, hb[f].v, (short)0, acc[j], false, false);
      }
    #pragma unroll
    for (int j = 8; j < NTILE; ++j)
      #pragma unroll
      for (int f = 0; f < 2; ++f) {
        const Frag16* ap = (const Frag16*)&ufr[((j * 2 + f) * 32 + lane) * 16];
        acz[j - 8] = __builtin_amdgcn_wmma_f32_16x16x32_f16(
            false, ap->v, false, hb[f].v, (short)0, acz[j - 8], false, false);
      }

    // ---- gates (pure per-lane VALU; co-executes with XDL WMMA) ----
    const float a_s  = abase[t];
    const bool valid = (t < sl);
    uint32_t pk[4][4];
    #pragma unroll
    for (int j = 0; j < 4; ++j) {
      float ov[8];
      #pragma unroll
      for (int r = 0; r < 8; ++r) {
        float ug = a_s * fast_sigmoid(acc[j][r]);          // includes bu
        float rr = fast_sigmoid(acc[4 + j][r]);            // includes br
        float z  = fast_tanh(acc[8 + j][r] + acz[j][r] * rr); // i_z(+bh) + h_z*r
        float hp = hst[j][r];
        float hn = hp + ug * (z - hp);
        hn = valid ? hn : hp;
        hst[j][r] = hn;
        ov[r] = valid ? hn : 0.0f;
      }
      float* op = obase + t * DD + 16 * j;
      *(float4*)(op)     = make_float4(ov[0], ov[1], ov[2], ov[3]);
      *(float4*)(op + 4) = make_float4(ov[4], ov[5], ov[6], ov[7]);
      #pragma unroll
      for (int q = 0; q < 4; ++q) {
        auto p = __builtin_amdgcn_cvt_pkrtz(hst[j][2 * q], hst[j][2 * q + 1]);
        pk[j][q] = __builtin_bit_cast(uint32_t, p);
      }
    }

    // ---- rebuild h^T B-fragments: half-swap with lane^16 via ds_swizzle SWAPX16 ----
    uint32_t xp2[4][4];
    #pragma unroll
    for (int j = 0; j < 4; ++j)
      #pragma unroll
      for (int q = 0; q < 4; ++q)
        xp2[j][q] = (uint32_t)__builtin_amdgcn_ds_swizzle((int)pk[j][q], 0x401F);
    const bool lo = (hi == 0);
    #pragma unroll
    for (int q = 0; q < 4; ++q) {
      hb[0].u[q]     = lo ? pk[0][q]  : xp2[1][q]; // frag0 e0..7 : K = 16*hi + 0..7
      hb[0].u[4 + q] = lo ? xp2[0][q] : pk[1][q];  // frag0 e8..15: K = 16*hi + 8..15
      hb[1].u[q]     = lo ? pk[2][q]  : xp2[3][q]; // frag1: K = 32 + 16*hi + ...
      hb[1].u[4 + q] = lo ? xp2[2][q] : pk[3][q];
    }
  }
}

extern "C" void kernel_launch(void* const* d_in, const int* in_sizes, int n_in,
                              void* d_out, int out_size, void* d_ws, size_t ws_size,
                              hipStream_t stream) {
  const float* x   = (const float*)d_in[0];
  const float* att = (const float*)d_in[1];
  const int*   sl  = (const int*)  d_in[2];
  const float* w   = (const float*)d_in[3];
  const float* u   = (const float*)d_in[4];
  const float* bu  = (const float*)d_in[5];
  const float* br  = (const float*)d_in[6];
  const float* bh  = (const float*)d_in[7];
  float* out = (float*)d_out;
  (void)in_sizes; (void)n_in; (void)out_size; (void)d_ws; (void)ws_size;

  dim3 grid(4096 / 16);   // one wave per 16-batch tile; recurrence bounds parallelism
  augru_scan_kernel<<<grid, 32, 0, stream>>>(x, att, sl, w, u, bu, br, bh, out);
}
